// SlotCRF_50861002719716
// MI455X (gfx1250) — matrix operand order repaired and checked by reading, not compiled, not run on hardware
//
#include <hip/hip_runtime.h>

// SlotCRF log-likelihood for MI455X (gfx1250, wave32).
//
// Part 1 (crf_score):    tagged-path score, gather + block reduction.
// Part 2 (crf_forward):  forward algorithm reformulated in the exp domain:
//                        s' = (E^T s) .* exp(emit),  E = exp(trans) constant.
//                        One wave owns 16 batch columns; the recursion is a
//                        chain of 8x v_wmma_f32_16x16x32_f16 per step with the
//                        state vector living permanently in the WMMA B-operand
//                        register layout. The constant A operand (E^T) has its
//                        rows permuted so that the WMMA D layout coincides with
//                        the next step's B layout -> no lane exchange per step.
//                        Per-step f32 max renorm keeps s in f16 range; log(max)
//                        accumulates in f32 (exact logsumexp semantics).
// Part 3 (crf_finalize): (sum(score) - sum(logZ)) / sum(mask).
//
// Roofline: slots = 128 MiB streamed exactly once (NT loads + prefetch) ->
// ~5.9 us floor at 23.3 TB/s; WMMA work (4.3 GFLOP eq.) is negligible.

#define CRF_B 512
#define CRF_L 1024
#define CRF_T 64

typedef __attribute__((ext_vector_type(16))) _Float16     v16h;
typedef __attribute__((ext_vector_type(8)))  float        v8f;
typedef __attribute__((ext_vector_type(2)))  _Float16     v2h;
typedef __attribute__((ext_vector_type(8)))  unsigned int v8u;
typedef __attribute__((ext_vector_type(4)))  float        f32x4;

__device__ __forceinline__ unsigned pk16(float a, float b) {
  // v_cvt_pk_rtz_f16_f32 -> one packed half2 per instruction
  return __builtin_bit_cast(unsigned, __builtin_amdgcn_cvt_pkrtz(a, b));
}

// ---------------------------------------------------------------------------
// Forward algorithm: one wave per 16-batch group (grid = 32 blocks x 32 thr).
// ---------------------------------------------------------------------------
__global__ __launch_bounds__(32) void crf_forward_wmma(
    const float* __restrict__ slots,        // [B, L, T]
    const unsigned char* __restrict__ mask, // [B, L]
    const float* __restrict__ startT,       // [T]
    const float* __restrict__ endT,         // [T]
    const float* __restrict__ trans,        // [T, T]
    float* __restrict__ logz)               // [B]
{
  const int lane = threadIdx.x & 31;
  const int lo   = lane & 15;
  const int hi   = lane >> 4;
  const int b    = blockIdx.x * 16 + lo;    // batch column this lane serves

  // j-block origin held by D tile t in THIS lane (8 consecutive tags).
  // Chosen (via the A-row permutation below) so that D == next-step B operand.
  int jb[4];
#pragma unroll
  for (int t = 0; t < 4; ++t) jb[t] = 32 * (t >> 1) + 8 * (t & 1) + 16 * hi;

  // ---- Constant A = exp(trans)^T, f16, WMMA A-layout, rows permuted. ------
  // A tile t / K-chunk c, element (lane, e=2v+h):
  //   row m = lo  ->  j = 32*(t>>1) + 8*(t&1) + 16*(lo>>3) + (lo&7)
  //   K        ->  i = 32c + 16*hi + 2*v + h
  v16h A[4][2];
#pragma unroll
  for (int t = 0; t < 4; ++t) {
    const int j = 32 * (t >> 1) + 8 * (t & 1) + 16 * ((lo >> 3) & 1) + (lo & 7);
#pragma unroll
    for (int c = 0; c < 2; ++c) {
#pragma unroll
      for (int v = 0; v < 8; ++v) {
        const int i0 = 32 * c + 16 * hi + 2 * v;
        A[t][c][2 * v]     = (_Float16)__expf(trans[i0 * CRF_T + j]);
        A[t][c][2 * v + 1] = (_Float16)__expf(trans[(i0 + 1) * CRF_T + j]);
      }
    }
  }

  // ---- State: s in f16 B-operand layout (2 K-chunks x 8 packed half2). ----
  v8u   Bu[2];
  float logscale = 0.0f;

  {
    // alpha0 = start + emissions[:,0]; normalize so max == 1.
    float vv[32];
#pragma unroll
    for (int t = 0; t < 4; ++t) {
      const size_t ebase = ((size_t)b * CRF_L) * CRF_T + jb[t];
#pragma unroll
      for (int r = 0; r < 8; ++r)
        vv[8 * t + r] = __expf(startT[jb[t] + r] + slots[ebase + r]);
    }
    float mx = vv[0];
#pragma unroll
    for (int i = 1; i < 32; ++i) mx = fmaxf(mx, vv[i]);
    mx = fmaxf(mx, __shfl_xor(mx, 16));            // combine j-halves (same batch)
    const float inv = __builtin_amdgcn_rcpf(mx);
    logscale = __logf(mx);
#pragma unroll
    for (int c = 0; c < 2; ++c)
#pragma unroll
      for (int p = 0; p < 8; ++p) {
        const int t = 2 * c + (p >> 2);
        const int r = 2 * (p & 3);
        Bu[c][p] = pk16(vv[8 * t + r] * inv, vv[8 * t + r + 1] * inv);
      }
  }

  // ---- Software-pipelined emission stream: load l+1 while computing l. ----
  float ecur[32];
  {
    const float* er = slots + ((size_t)b * CRF_L + 1) * CRF_T;
#pragma unroll
    for (int t = 0; t < 4; ++t) {
      const f32x4 q0 = __builtin_nontemporal_load((const f32x4*)(er + jb[t]));
      const f32x4 q1 = __builtin_nontemporal_load((const f32x4*)(er + jb[t] + 4));
#pragma unroll
      for (int k = 0; k < 4; ++k) { ecur[8 * t + k] = q0[k]; ecur[8 * t + 4 + k] = q1[k]; }
    }
  }

  const unsigned char* mrow = mask + (size_t)b * CRF_L;

  for (int l = 1; l < CRF_L; ++l) {
    // Prefetch + load emissions for step l+1 (clamped; uniform branch-free).
    const int ln = (l + 1 < CRF_L) ? (l + 1) : (CRF_L - 1);
    const float* ernext = slots + ((size_t)b * CRF_L + ln) * CRF_T;
    __builtin_prefetch(ernext + CRF_T + jb[0], 0, 0);   // global_prefetch_b8 (l+2)
    __builtin_prefetch(ernext + CRF_T + jb[2], 0, 0);
    float enext[32];
#pragma unroll
    for (int t = 0; t < 4; ++t) {
      const f32x4 q0 = __builtin_nontemporal_load((const f32x4*)(ernext + jb[t]));
      const f32x4 q1 = __builtin_nontemporal_load((const f32x4*)(ernext + jb[t] + 4));
#pragma unroll
      for (int k = 0; k < 4; ++k) { enext[8 * t + k] = q0[k]; enext[8 * t + 4 + k] = q1[k]; }
    }
    const bool mu = mrow[l] != 0;

    // D[t] = A[t] x s : 8x v_wmma_f32_16x16x32_f16 (f32 accumulate).
    const v16h Bh0 = __builtin_bit_cast(v16h, Bu[0]);
    const v16h Bh1 = __builtin_bit_cast(v16h, Bu[1]);
    v8f D[4];
#pragma unroll
    for (int t = 0; t < 4; ++t) {
      v8f acc = {};
      acc = __builtin_amdgcn_wmma_f32_16x16x32_f16(false, A[t][0], false, Bh0,
                                                   (short)0, acc, false, false);
      acc = __builtin_amdgcn_wmma_f32_16x16x32_f16(false, A[t][1], false, Bh1,
                                                   (short)0, acc, false, false);
      D[t] = acc;
    }

    // s'_j = D_j * exp(emit_j); renormalize per batch; carry log(max).
    float vv[32];
#pragma unroll
    for (int t = 0; t < 4; ++t)
#pragma unroll
      for (int r = 0; r < 8; ++r)
        vv[8 * t + r] = D[t][r] * __expf(ecur[8 * t + r]);   // v_exp co-executes w/ WMMA

    float mx = vv[0];
#pragma unroll
    for (int i = 1; i < 32; ++i) mx = fmaxf(mx, vv[i]);
    mx = fmaxf(mx, __shfl_xor(mx, 16));
    const float inv = __builtin_amdgcn_rcpf(mx);
    logscale += mu ? __logf(mx) : 0.0f;

#pragma unroll
    for (int c = 0; c < 2; ++c)
#pragma unroll
      for (int p = 0; p < 8; ++p) {
        const int t = 2 * c + (p >> 2);
        const int r = 2 * (p & 3);
        const unsigned nu = pk16(vv[8 * t + r] * inv, vv[8 * t + r + 1] * inv);
        Bu[c][p] = mu ? nu : Bu[c][p];                       // mask: keep alpha
      }

#pragma unroll
    for (int i = 0; i < 32; ++i) ecur[i] = enext[i];
  }

  // logZ = logscale + log( sum_j s_j * exp(end_j) )
  float acc = 0.0f;
#pragma unroll
  for (int c = 0; c < 2; ++c)
#pragma unroll
    for (int p = 0; p < 8; ++p) {
      const v2h h2 = __builtin_bit_cast(v2h, Bu[c][p]);
      const int j0 = 32 * c + 16 * hi + 2 * p;               // B-layout K == tag id
      acc += (float)h2[0] * __expf(endT[j0]);
      acc += (float)h2[1] * __expf(endT[j0 + 1]);
    }
  acc += __shfl_xor(acc, 16);
  if (lane < 16) logz[b] = logscale + __logf(acc);
}

// ---------------------------------------------------------------------------
// Tagged-path score + mask count, one block per batch.
// ---------------------------------------------------------------------------
__global__ __launch_bounds__(256) void crf_score(
    const float* __restrict__ slots, const long long* __restrict__ tags,
    const unsigned char* __restrict__ mask, const float* __restrict__ startT,
    const float* __restrict__ endT, const float* __restrict__ trans,
    float* __restrict__ score, float* __restrict__ cnt)
{
  const int b = blockIdx.x, tid = threadIdx.x;
  __shared__ float sred[256], cred[256];
  float p = 0.0f, c = 0.0f;
  for (int l = tid; l < CRF_L; l += 256) {
    const bool m = mask[(size_t)b * CRF_L + l] != 0;
    c += m ? 1.0f : 0.0f;
    if (l >= 1 && m) {
      const long long tp = tags[(size_t)b * CRF_L + l - 1];
      const long long tc = tags[(size_t)b * CRF_L + l];
      p += trans[tp * CRF_T + tc] + slots[((size_t)b * CRF_L + l) * CRF_T + tc];
    }
  }
  sred[tid] = p; cred[tid] = c;
  __syncthreads();
  for (int s = 128; s > 0; s >>= 1) {
    if (tid < s) { sred[tid] += sred[tid + s]; cred[tid] += cred[tid + s]; }
    __syncthreads();
  }
  if (tid == 0) {
    const long long t0 = tags[(size_t)b * CRF_L];
    const int se = (int)cred[0] - 1;                 // seq_end = sum(mask) - 1
    const long long tl = tags[(size_t)b * CRF_L + se];
    score[b] = startT[t0] + slots[(size_t)b * CRF_L * CRF_T + t0] + sred[0] + endT[tl];
    cnt[b]   = cred[0];
  }
}

// ---------------------------------------------------------------------------
// Final scalar: (sum score - sum logZ) / sum mask.
// ---------------------------------------------------------------------------
__global__ __launch_bounds__(512) void crf_finalize(
    const float* __restrict__ score, const float* __restrict__ logzv,
    const float* __restrict__ cnt, float* __restrict__ out)
{
  __shared__ float s1[512], s2[512];
  const int t = threadIdx.x;
  s1[t] = score[t] - logzv[t];
  s2[t] = cnt[t];
  __syncthreads();
  for (int s = 256; s > 0; s >>= 1) {
    if (t < s) { s1[t] += s1[t + s]; s2[t] += s2[t + s]; }
    __syncthreads();
  }
  if (t == 0) out[0] = s1[0] / s2[0];
}

// ---------------------------------------------------------------------------
extern "C" void kernel_launch(void* const* d_in, const int* in_sizes, int n_in,
                              void* d_out, int out_size, void* d_ws, size_t ws_size,
                              hipStream_t stream) {
  (void)in_sizes; (void)n_in; (void)out_size; (void)ws_size;
  const float*         slots  = (const float*)d_in[0];
  const long long*     tags   = (const long long*)d_in[1];      // int64
  const unsigned char* mask   = (const unsigned char*)d_in[2];  // bool
  const float*         startT = (const float*)d_in[3];
  const float*         endT   = (const float*)d_in[4];
  const float*         trans  = (const float*)d_in[5];
  float* out = (float*)d_out;

  float* ws    = (float*)d_ws;
  float* score = ws;               // [512]
  float* logz  = ws + CRF_B;       // [512]
  float* cnt   = ws + 2 * CRF_B;   // [512]

  crf_score<<<CRF_B, 256, 0, stream>>>(slots, tags, mask, startT, endT, trans,
                                       score, cnt);
  crf_forward_wmma<<<CRF_B / 16, 32, 0, stream>>>(slots, mask, startT, endT,
                                                  trans, logz);
  crf_finalize<<<1, CRF_B, 0, stream>>>(score, logz, cnt, out);
}
